// Embedding_44418551775446
// MI455X (gfx1250) — matrix operand order repaired and checked by reading, not compiled, not run on hardware
//
#include <hip/hip_runtime.h>

typedef __attribute__((ext_vector_type(16))) _Float16 v16h;
typedef __attribute__((ext_vector_type(8)))  _Float16 v8h;
typedef __attribute__((ext_vector_type(8)))  float    v8f;
typedef __attribute__((ext_vector_type(4)))  float    v4f;

#define kBS   16
#define kSL   2048
#define kW    16
#define kDR   64
#define kDW   64
#define kH    256
#define kHH   128
#define kG4   512          // 4*HH
#define kNTok (kBS * kSL)  // 32768
#define kWaves 4
#define kTPB   (kWaves * 32)
#define kLnEps 1e-5f

// ---- LDS layout (bytes) ----
#define OFF_WIH 0                                   // 512*64  f16 = 64KB  (phase3: Wc f16 128KB reuses WIH+half of WHH)
#define OFF_WHH (64 * 1024)                         // 512*128 f16 = 128KB
#define OFF_CONCAT (192 * 1024)                     // per-wave 16*256 f16 = 8KB
#define OFF_BWR (OFF_CONCAT + kWaves * 16 * 256 * 2)// per-wave 16*256 f32 = 16KB
#define SMEM_BYTES (OFF_BWR + kWaves * 16 * 256 * 4)// 294912 = 288KB

// overflow-safe fast activations (single v_exp_f32 each)
__device__ __forceinline__ float sigm(float x) { return 1.0f / (1.0f + __expf(-x)); }
__device__ __forceinline__ float tanh_fast(float x) {
  float e = __expf(2.0f * x);          // inf -> 1, 0 -> -1 : no NaN at extremes
  return 1.0f - 2.0f / (e + 1.0f);
}

__device__ __forceinline__ v8f wmma_f16(v16h a, v16h b, v8f c) {
  return __builtin_amdgcn_wmma_f32_16x16x32_f16(false, a, false, b, (short)0, c, false, false);
}

// A 16x32 f16 fragment from row-major f32 global; lane (l&15)=row, (l>>4) selects K sub-run.
// Contiguous: element j -> K = k0 + sub*16 + j (matches B loaders)
__device__ __forceinline__ v16h load_a_f32g(const float* src, int ld, int k0, int lane) {
  int row = lane & 15, sub = lane >> 4;
  const float* p0 = src + (size_t)row * ld + k0 + sub * 16;
  const float* p1 = p0 + 8;
  v16h a;
#pragma unroll
  for (int j = 0; j < 8; ++j) { a[j] = (_Float16)p0[j]; a[8 + j] = (_Float16)p1[j]; }
  return a;
}

// Same A layout but from f16 LDS (16B-aligned runs).
__device__ __forceinline__ v16h load_a_lds(const _Float16* src, int ld, int k0, int lane) {
  int row = lane & 15, sub = lane >> 4;
  const _Float16* p = src + row * ld + k0 + sub * 16;
  union { v16h v; v8h h[2]; } u;
  u.h[0] = *(const v8h*)p;
  u.h[1] = *(const v8h*)(p + 8);
  return u.v;
}

// B 32x16 f16 fragment: B[k][n] = Wt[n][k] (Wt row-major [Nout][K] in LDS f16).
// lane halves [0..15] -> K = k0 + (l>>4)*16 + j, col n = n0 + (l&15); run is contiguous in K.
__device__ __forceinline__ v16h load_b_lds(const _Float16* wt, int ld, int n0, int k0, int lane) {
  int col = lane & 15, sub = lane >> 4;
  const _Float16* p = wt + (n0 + col) * ld + k0 + sub * 16;
  union { v16h v; v8h h[2]; } u;
  u.h[0] = *(const v8h*)p;
  u.h[1] = *(const v8h*)(p + 8);
  return u.v;
}

// B fragment gathered from row-major f32 global with on-the-fly f32->f16 convert.
__device__ __forceinline__ v16h load_b_f32g(const float* wt, int ld, int n0, int k0, int lane) {
  int col = lane & 15, sub = lane >> 4;
  const float* p = wt + (size_t)(n0 + col) * ld + k0 + sub * 16;
  v16h b;
#pragma unroll
  for (int j = 0; j < 16; ++j) b[j] = (_Float16)p[j];
  return b;
}

__global__ void fused_lstm_ln_kernel(
    const float* __restrict__ xr, const float* __restrict__ xw, const long long* __restrict__ xn,
    const float* __restrict__ Wr_w, const float* __restrict__ Wr_b,
    const float* __restrict__ W_ih, const float* __restrict__ W_hh,
    const float* __restrict__ b_ih, const float* __restrict__ b_hh,
    const float* __restrict__ Wc_w, const float* __restrict__ Wc_b,
    const float* __restrict__ ln_g, const float* __restrict__ ln_b,
    float* __restrict__ out) {
  extern __shared__ __align__(256) char smem[];
  _Float16* lds_wih = (_Float16*)(smem + OFF_WIH);   // [512][64]
  _Float16* lds_whh = (_Float16*)(smem + OFF_WHH);   // [512][128]
  _Float16* lds_wc  = (_Float16*)(smem + OFF_WIH);   // [256][256] (phase 3)

  const int wave = threadIdx.x >> 5;
  const int lane = threadIdx.x & 31;
  const int col  = lane & 15;
  const int sub  = lane >> 4;

  _Float16* concat = (_Float16*)(smem + OFF_CONCAT) + wave * (16 * 256); // [16][256] f16
  float*    bwr    = (float*)(smem + OFF_BWR) + wave * (16 * 256);        // [16][256] f32

  // ---- cooperative weight stage: f32 -> f16 into LDS ----
  for (int i = threadIdx.x; i < kG4 * kDW; i += blockDim.x) lds_wih[i] = (_Float16)W_ih[i];
  for (int i = threadIdx.x; i < kG4 * kHH; i += blockDim.x) lds_whh[i] = (_Float16)W_hh[i];
  __syncthreads();

  const int tile = blockIdx.x * kWaves + wave;  // 16-token tile
  const int tok0 = tile * 16;

  // ---- phase 1: br = relu(xr @ Wr_w^T + Wr_b) -> concat cols [0,128) as f16 ----
  {
    v16h ax0 = load_a_f32g(xr + (size_t)tok0 * kDR, kDR, 0, lane);
    v16h ax1 = load_a_f32g(xr + (size_t)tok0 * kDR, kDR, 32, lane);
#pragma unroll
    for (int jt = 0; jt < 8; ++jt) {
      v8f acc = {};  // inline-0 C operand
      acc = wmma_f16(ax0, load_b_f32g(Wr_w, kDR, jt * 16, 0, lane), acc);
      acc = wmma_f16(ax1, load_b_f32g(Wr_w, kDR, jt * 16, 32, lane), acc);
      const float bv = Wr_b[jt * 16 + col];
#pragma unroll
      for (int r = 0; r < 8; ++r) {
        float v = acc[r] + bv;
        v = v > 0.0f ? v : 0.0f;
        concat[(sub * 8 + r) * 256 + jt * 16 + col] = (_Float16)v;
      }
    }
  }
  // zero h region (concat cols [128,256))
  for (int i = lane; i < 16 * 128; i += 32)
    concat[(i >> 7) * 256 + 128 + (i & 127)] = (_Float16)0.0f;

  // ragged lengths for the 8 C-rows this lane touches
  int lens8[8];
#pragma unroll
  for (int r = 0; r < 8; ++r) lens8[r] = (int)xn[(size_t)(tok0 + sub * 8 + r) * 5 + 4];

  // per-lane combined gate bias, hoisted out of the recurrence (statically indexed)
  float bias_c[4][8];
#pragma unroll
  for (int gi = 0; gi < 4; ++gi)
#pragma unroll
    for (int j = 0; j < 8; ++j) {
      const int n = gi * 128 + j * 16 + col;
      bias_c[gi][j] = b_ih[n] + b_hh[n];
    }

  // persistent cell state, 8 column tiles of 16x16 (f32 C-layout)
  v8f cst[8];
#pragma unroll
  for (int j = 0; j < 8; ++j) cst[j] = (v8f){};

  // ---- phase 2: 16-step ragged LSTM ----
  const float* xw_base = xw + (size_t)tok0 * (kW * kDW);
#pragma unroll 1
  for (int t = 0; t < kW; ++t) {
    asm volatile("s_wait_dscnt 0" ::: "memory");  // h stores from step t-1 visible
    v16h axw0 = load_a_f32g(xw_base + t * kDW, kW * kDW, 0, lane);
    v16h axw1 = load_a_f32g(xw_base + t * kDW, kW * kDW, 32, lane);
    v16h ah[4];
#pragma unroll
    for (int kk = 0; kk < 4; ++kk) ah[kk] = load_a_lds(concat, 256, 128 + kk * 32, lane);

#pragma unroll
    for (int j = 0; j < 8; ++j) {
      v8f g[4];
#pragma unroll
      for (int gi = 0; gi < 4; ++gi) {
        const int n0 = gi * 128 + j * 16;
        v8f acc = {};  // inline-0 C operand; bias folded into the gate epilogue
        acc = wmma_f16(axw0, load_b_lds(lds_wih, kDW, n0, 0, lane), acc);
        acc = wmma_f16(axw1, load_b_lds(lds_wih, kDW, n0, 32, lane), acc);
#pragma unroll
        for (int kk = 0; kk < 4; ++kk)
          acc = wmma_f16(ah[kk], load_b_lds(lds_whh, kHH, n0, kk * 32, lane), acc);
        g[gi] = acc;
      }
      // gate math + ragged freeze (mask per C-row)
#pragma unroll
      for (int r = 0; r < 8; ++r) {
        if (t < lens8[r]) {
          float iv = sigm(g[0][r] + bias_c[0][j]);
          float fv = sigm(g[1][r] + bias_c[1][j]);
          float gv = tanh_fast(g[2][r] + bias_c[2][j]);
          float ov = sigm(g[3][r] + bias_c[3][j]);
          float cn = fv * cst[j][r] + iv * gv;
          cst[j][r] = cn;
          float hn = ov * tanh_fast(cn);
          concat[(sub * 8 + r) * 256 + 128 + j * 16 + col] = (_Float16)hn;
        }
      }
    }
  }

  // ---- phase 3: bwr = concat(br,bw) @ Wc^T + Wc_b ; then LayerNorm ----
  __syncthreads();  // everyone done with W_ih/W_hh
  for (int i = threadIdx.x; i < kH * kH; i += blockDim.x) lds_wc[i] = (_Float16)Wc_w[i];
  __syncthreads();

  {
    v16h af[8];
#pragma unroll
    for (int kk = 0; kk < 8; ++kk) af[kk] = load_a_lds(concat, 256, kk * 32, lane);
#pragma unroll 1
    for (int jt = 0; jt < 16; ++jt) {
      v8f acc = {};
#pragma unroll
      for (int kk = 0; kk < 8; ++kk)
        acc = wmma_f16(af[kk], load_b_lds(lds_wc, kH, jt * 16, kk * 32, lane), acc);
      const float bv = Wc_b[jt * 16 + col];
#pragma unroll
      for (int r = 0; r < 8; ++r)
        bwr[(sub * 8 + r) * 256 + jt * 16 + col] = acc[r] + bv;
    }
  }
  asm volatile("s_wait_dscnt 0" ::: "memory");

  // LayerNorm over H=256: 2 lanes per row, shfl-combine, fused affine + store
  {
    const int r2 = lane >> 1;       // row 0..15
    const int hv = lane & 1;        // column half
    const float* prow = bwr + r2 * 256 + hv * 128;
    float s = 0.0f, s2 = 0.0f;
#pragma unroll
    for (int i = 0; i < 128; i += 4) {
      v4f v = *(const v4f*)(prow + i);
      s  += v[0] + v[1] + v[2] + v[3];
      s2 += v[0] * v[0] + v[1] * v[1] + v[2] * v[2] + v[3] * v[3];
    }
    s  += __shfl_xor(s, 1, 32);
    s2 += __shfl_xor(s2, 1, 32);
    const float mu  = s * (1.0f / 256.0f);
    const float var = s2 * (1.0f / 256.0f) - mu * mu;
    const float rs  = rsqrtf(var + kLnEps);
    float* po = out + (size_t)(tok0 + r2) * kH + hv * 128;
    const float* pg = ln_g + hv * 128;
    const float* pb = ln_b + hv * 128;
#pragma unroll 4
    for (int i = 0; i < 128; ++i)
      po[i] = (prow[i] - mu) * rs * pg[i] + pb[i];
  }
}

extern "C" void kernel_launch(void* const* d_in, const int* in_sizes, int n_in,
                              void* d_out, int out_size, void* d_ws, size_t ws_size,
                              hipStream_t stream) {
  const float*     xr   = (const float*)d_in[0];
  const float*     xw   = (const float*)d_in[1];
  const long long* xn   = (const long long*)d_in[2];
  const float*     Wr_w = (const float*)d_in[3];
  const float*     Wr_b = (const float*)d_in[4];
  const float*     W_ih = (const float*)d_in[5];
  const float*     W_hh = (const float*)d_in[6];
  const float*     b_ih = (const float*)d_in[7];
  const float*     b_hh = (const float*)d_in[8];
  const float*     Wc_w = (const float*)d_in[9];
  const float*     Wc_b = (const float*)d_in[10];
  const float*     ln_g = (const float*)d_in[11];
  const float*     ln_b = (const float*)d_in[12];
  float* out = (float*)d_out;

  (void)hipFuncSetAttribute(reinterpret_cast<const void*>(fused_lstm_ln_kernel),
                            hipFuncAttributeMaxDynamicSharedMemorySize, SMEM_BYTES);
  dim3 grid(kNTok / (16 * kWaves));  // 512 blocks, 4 waves each, 16 tokens per wave
  fused_lstm_ln_kernel<<<grid, kTPB, SMEM_BYTES, stream>>>(
      xr, xw, xn, Wr_w, Wr_b, W_ih, W_hh, b_ih, b_hh, Wc_w, Wc_b, ln_g, ln_b, out);
}
